// DepthHead_5695126634905
// MI455X (gfx1250) — matrix operand order repaired and checked by reading, not compiled, not run on hardware
//
#include <hip/hip_runtime.h>
#include <hip/hip_bf16.h>

// ---------------------------------------------------------------------------
// Types for CDNA5 WMMA (gfx1250, wave32): D(16x16 f32) = A(16x32 bf16) x B(32x16 bf16) + C
// ---------------------------------------------------------------------------
typedef __attribute__((ext_vector_type(16))) __bf16 v16b;
typedef __attribute__((ext_vector_type(8)))  __bf16 v8b;
typedef __attribute__((ext_vector_type(8)))  float  v8f;
typedef __attribute__((ext_vector_type(4)))  unsigned int u32x4;
typedef __attribute__((ext_vector_type(8)))  int i32x8;
typedef __attribute__((ext_vector_type(4)))  int i32x4;

#define WAVES 4       // 4 waves (128 threads) per block; each wave owns a 16x16 D tile
#define MAX_KPAD 640  // max padded K (covers 608)

// ---------------------------------------------------------------------------
// Fragment helpers.
// A-matrix 16x32 bf16 (ISA 7.12.2): lane L -> row m=L&15, half h=L>>4,
//   element j -> k = kk + 16*h + j (contiguous, not ISA table order)  => two contiguous 8-elem chunks
//   at k = kk+16h and k = kk+16h+8.
// B-matrix 32x16 bf16: lane L -> col n=L&15, holds k = kk+16h .. kk+16h+15.
// ---------------------------------------------------------------------------
__device__ __forceinline__ void ld_chunk_f32(v16b& a, int off, const float* __restrict__ p) {
#pragma unroll
  for (int j = 0; j < 8; ++j) a[off + j] = (__bf16)p[j];
}
__device__ __forceinline__ void zero_chunk(v16b& a, int off) {
#pragma unroll
  for (int j = 0; j < 8; ++j) a[off + j] = (__bf16)0.0f;
}
__device__ __forceinline__ void ld_chunk_bf(v16b& a, int off, const __bf16* __restrict__ p) {
  v8b v = *(const v8b*)p;
#pragma unroll
  for (int j = 0; j < 8; ++j) a[off + j] = v[j];
}

// ---------------------------------------------------------------------------
// TDM: DMA a contiguous run of `elems` bf16 weights into LDS offset `lds_off`.
// D# per CDNA5 ISA 8.3/8.4: group0 {count=1, lds_addr, global_addr, type=2},
// group1 {data_size=2B, tensor_dim0=tile_dim0=elems, tensor_dim1=1,
// tensor_dim0_stride=elems}. 1-D tile: tile_dim1/2=0. Tracked by TENSORcnt.
// ---------------------------------------------------------------------------
__device__ __forceinline__ void tdm_load_wt(const __bf16* gsrc, unsigned lds_off, int elems) {
  unsigned long long ga = (unsigned long long)(size_t)gsrc;
  u32x4 g0;
  g0[0] = 1u;                                            // count=1 (valid user D#)
  g0[1] = lds_off;                                       // lds_addr (bytes)
  g0[2] = (unsigned)(ga & 0xFFFFFFFFu);                  // global_addr[31:0]
  g0[3] = (unsigned)((ga >> 32) & 0x1FFFFFFu) | (2u << 30);  // addr[56:32] | type=2
  unsigned e = (unsigned)elems;
  i32x8 g1;
  g1[0] = 0x00010000;                       // workgroup_mask=0, data_size=1 (2 bytes)
  g1[1] = (int)((e & 0xFFFFu) << 16);       // tensor_dim0[15:0]
  g1[2] = (int)((e >> 16) | (1u << 16));    // tensor_dim0[31:16], tensor_dim1=1
  g1[3] = (int)((e & 0xFFFFu) << 16);       // tile_dim0 = elems
  g1[4] = 0;                                // tile_dim1 = tile_dim2 = 0 (unused)
  g1[5] = (int)e;                           // tensor_dim0_stride[31:0]
  g1[6] = 0;
  g1[7] = 0;
  i32x4 z4 = {0, 0, 0, 0};
#if __clang_major__ >= 23
  i32x8 z8 = {0, 0, 0, 0, 0, 0, 0, 0};
  __builtin_amdgcn_tensor_load_to_lds(g0, g1, z4, z4, z8, 0);
#else
  __builtin_amdgcn_tensor_load_to_lds(g0, g1, z4, z4, 0);
#endif
}

// The TDM write to LDS is invisible to LLVM's alias analysis; without this
// fence the compiler folds reads of the (apparently never-stored) shared tile
// to undef and deletes the ds loads. The escaped pointer + memory clobber
// forces the compiler to assume the tile was written.
__device__ __forceinline__ void lds_publish(const void* p) {
  asm volatile("" : : "v"(p) : "memory");
}

// ---------------------------------------------------------------------------
// Generic row-GEMM: out[M,Cout] = epi( A[M,Kpad] @ Wt^T ), A bf16 row-major,
// Wt bf16 [Cout, Kpad] (k-contiguous). The 16xKpad B tile for this block's
// column tile is DMA'd into LDS by the TDM, then read via ds loads.
// Epilogue: (acc+b)*g+be -> relu -> +res.  M multiple of 16*WAVES.
// ---------------------------------------------------------------------------
__global__ void wmma_gemm_rows(const __bf16* __restrict__ A, int Kpad,
                               const __bf16* __restrict__ Wt,
                               const float* __restrict__ bias,
                               const float* __restrict__ gam,
                               const float* __restrict__ bet,
                               const float* __restrict__ res,
                               float* __restrict__ out, int Cout) {
  __shared__ __bf16 lwt[16 * MAX_KPAD];  // at LDS offset 0 (single shared object)
  const int lane = threadIdx.x & 31;
  const int wv   = threadIdx.x >> 5;
  if (wv == 0) {  // wave-uniform: one TDM transfer per workgroup
    tdm_load_wt(Wt + (size_t)blockIdx.y * 16 * Kpad, 0u, 16 * Kpad);
    __builtin_amdgcn_s_wait_tensorcnt(0);
  }
  __syncthreads();
  lds_publish(lwt);

  const int row0 = (blockIdx.x * WAVES + wv) * 16;
  const int m    = lane & 15;
  const int h    = lane >> 4;
  const __bf16* arow = A + (size_t)(row0 + m) * Kpad;
  const __bf16* brow = lwt + (size_t)m * Kpad;
  v8f acc = {};
  for (int kk = 0; kk < Kpad; kk += 32) {
    v16b a, b;
    ld_chunk_bf(a, 0, arow + kk + 16 * h);
    ld_chunk_bf(a, 8, arow + kk + 16 * h + 8);
    b = *(const v16b*)(brow + kk + 16 * h);
    __builtin_prefetch(arow + kk + 64, 0, 1);  // global_prefetch_b8 (speculative-safe)
    acc = __builtin_amdgcn_wmma_f32_16x16x32_bf16(false, a, false, b, (short)0, acc,
                                                  false, false);
  }
  const int n = blockIdx.y * 16 + m;  // D: lane -> column n
  if (n >= Cout) return;
  const float bb = bias[n];
  const float gg = gam ? gam[n] : 1.0f;
  const float ee = gam ? bet[n] : 0.0f;
#pragma unroll
  for (int v = 0; v < 8; ++v) {  // D: VGPR v -> row v + 8*h
    const int r = row0 + v + 8 * h;
    float val = fmaxf((acc[v] + bb) * gg + ee, 0.0f);
    const size_t idx = (size_t)r * Cout + n;
    if (res) val += res[idx];
    out[idx] = val;
  }
}

// ---------------------------------------------------------------------------
// Implicit-im2row 3x3 conv (padding=1, stride=1) as WMMA GEMM over pixels.
// K index: k = tap*Cin + ci.  Cin % 8 == 0 so each 8-elem chunk stays inside
// one tap. Source stays fp32 (shared with scatter/residual consumers).
// ---------------------------------------------------------------------------
__device__ __forceinline__ void conv_a_chunk(v16b& a, int off, const float* __restrict__ S,
                                             int py, int px, int H, int W, int C, int kc) {
  if (kc >= 9 * C) { zero_chunk(a, off); return; }
  const int tap = kc / C;
  const int ci  = kc - tap * C;
  const int qy  = py + tap / 3 - 1;
  const int qx  = px + tap % 3 - 1;
  if (qy < 0 || qy >= H || qx < 0 || qx >= W) { zero_chunk(a, off); return; }
  ld_chunk_f32(a, off, S + (size_t)(qy * W + qx) * C + ci);
}

__global__ void wmma_conv3x3(const float* __restrict__ S, int H, int W, int Cin,
                             const __bf16* __restrict__ Wt, int Kpad,
                             const float* __restrict__ bias,
                             const float* __restrict__ gam,
                             const float* __restrict__ bet,
                             const float* __restrict__ res,
                             float* __restrict__ out, int Cout) {
  __shared__ __bf16 lwt[16 * MAX_KPAD];
  const int lane = threadIdx.x & 31;
  const int wv   = threadIdx.x >> 5;
  if (wv == 0) {
    tdm_load_wt(Wt + (size_t)blockIdx.y * 16 * Kpad, 0u, 16 * Kpad);
    __builtin_amdgcn_s_wait_tensorcnt(0);
  }
  __syncthreads();
  lds_publish(lwt);

  const int row0 = (blockIdx.x * WAVES + wv) * 16;
  const int m    = lane & 15;
  const int h    = lane >> 4;
  const int p    = row0 + m;
  const int py   = p / W;
  const int px   = p % W;
  const __bf16* brow = lwt + (size_t)m * Kpad;
  v8f acc = {};
  for (int kk = 0; kk < Kpad; kk += 32) {
    v16b a, b;
    conv_a_chunk(a, 0, S, py, px, H, W, Cin, kk + 16 * h);
    conv_a_chunk(a, 8, S, py, px, H, W, Cin, kk + 16 * h + 8);
    b = *(const v16b*)(brow + kk + 16 * h);
    acc = __builtin_amdgcn_wmma_f32_16x16x32_bf16(false, a, false, b, (short)0, acc,
                                                  false, false);
  }
  const int n = blockIdx.y * 16 + m;
  if (n >= Cout) return;
  const float bb = bias[n];
  const float gg = gam ? gam[n] : 1.0f;
  const float ee = gam ? bet[n] : 0.0f;
#pragma unroll
  for (int v = 0; v < 8; ++v) {
    const int r = row0 + v + 8 * h;
    float val = fmaxf((acc[v] + bb) * gg + ee, 0.0f);
    const size_t idx = (size_t)r * Cout + n;
    if (res) val += res[idx];
    out[idx] = val;
  }
}

// ---------------------------------------------------------------------------
// Weight repack kernels (fp32 -> bf16, zero-padded, k-contiguous per out-chan)
// ---------------------------------------------------------------------------
__global__ void cvt_wt_cc(__bf16* wt, const float* __restrict__ w, int Kdim, int Cout, int Kpad) {
  int idx = blockIdx.x * blockDim.x + threadIdx.x;
  if (idx >= Cout * Kpad) return;
  int n = idx / Kpad, k = idx - n * Kpad;
  wt[idx] = (k < Kdim) ? (__bf16)w[(size_t)k * Cout + n] : (__bf16)0.0f;
}
__global__ void cvt_wt_pw(__bf16* wt, const float* __restrict__ w, int Cin, int Kpad, int Cout) {
  int idx = blockIdx.x * blockDim.x + threadIdx.x;
  if (idx >= Cout * Kpad) return;
  int n = idx / Kpad, k = idx - n * Kpad;
  wt[idx] = (k < Cin) ? (__bf16)w[(size_t)n * Cin + k] : (__bf16)0.0f;
}
__global__ void cvt_wt_conv(__bf16* wt, const float* __restrict__ w, int Cin, int Kpad, int Cout) {
  int idx = blockIdx.x * blockDim.x + threadIdx.x;
  if (idx >= Cout * Kpad) return;
  int n = idx / Kpad, k = idx - n * Kpad;
  if (k < 9 * Cin) {
    int tap = k / Cin, ci = k - tap * Cin;
    wt[idx] = (__bf16)w[((size_t)n * Cin + ci) * 9 + tap];
  } else {
    wt[idx] = (__bf16)0.0f;
  }
}

// ---------------------------------------------------------------------------
// Direct small convs (stems / final 32->1), NHWC in/out, OIHW weights.
// ---------------------------------------------------------------------------
__global__ void direct_conv3x3(const float* __restrict__ in, int Hin, int Win, int Cin,
                               int stride, const float* __restrict__ w,
                               const float* __restrict__ b, const float* __restrict__ g,
                               const float* __restrict__ be, float* __restrict__ out,
                               int Hout, int Wout, int Cout) {
  int idx = blockIdx.x * blockDim.x + threadIdx.x;
  if (idx >= Hout * Wout * Cout) return;
  int co = idx % Cout;
  int p  = idx / Cout;
  int yo = p / Wout, xo = p % Wout;
  float acc = b[co];
  for (int ci = 0; ci < Cin; ++ci) {
#pragma unroll
    for (int t = 0; t < 9; ++t) {
      int yi = yo * stride + t / 3 - 1;
      int xi = xo * stride + t % 3 - 1;
      if (yi < 0 || yi >= Hin || xi < 0 || xi >= Win) continue;
      acc += w[((size_t)co * Cin + ci) * 9 + t] * in[(size_t)(yi * Win + xi) * Cin + ci];
    }
  }
  float val = g ? (acc * g[co] + be[co]) : acc;
  out[(size_t)p * Cout + co] = fmaxf(val, 0.0f);
}

// Depthwise 3x3 (+bias), NHWC fp32 in; bf16 out with stride Cs (zero-padded)
__global__ void dwconv3x3(const float* __restrict__ in, int Hin, int Win, int C, int stride,
                          const float* __restrict__ w, const float* __restrict__ b,
                          __bf16* __restrict__ out, int Hout, int Wout, int Cs) {
  int idx = blockIdx.x * blockDim.x + threadIdx.x;
  if (idx >= Hout * Wout * Cs) return;
  int c = idx % Cs;
  int p = idx / Cs;
  if (c >= C) { out[idx] = (__bf16)0.0f; return; }
  int yo = p / Wout, xo = p % Wout;
  float acc = b[c];
#pragma unroll
  for (int t = 0; t < 9; ++t) {
    int yi = yo * stride + t / 3 - 1;
    int xi = xo * stride + t % 3 - 1;
    if (yi < 0 || yi >= Hin || xi < 0 || xi >= Win) continue;
    acc += w[c * 9 + t] * in[(size_t)(yi * Win + xi) * C + c];
  }
  out[idx] = (__bf16)acc;
}

// ---------------------------------------------------------------------------
// Misc data-movement kernels
// ---------------------------------------------------------------------------
__global__ void pack_rgbd(float* __restrict__ o, const float* __restrict__ img,
                          const float* __restrict__ sd, int P) {
  int p = blockIdx.x * blockDim.x + threadIdx.x;
  if (p >= P) return;
  o[p * 4 + 0] = img[p];
  o[p * 4 + 1] = img[P + p];
  o[p * 4 + 2] = img[2 * P + p];
  o[p * 4 + 3] = sd[p];
}

__global__ void upcat(float* __restrict__ y, const float* __restrict__ yr,
                      const float* __restrict__ ys, int H, int W) {
  int idx = blockIdx.x * blockDim.x + threadIdx.x;
  if (idx >= H * W * 48) return;
  int c = idx % 48;
  int p = idx / 48;
  int ph = ((p / W) >> 1) * (W >> 1) + ((p % W) >> 1);
  y[idx] = (c < 32) ? yr[(size_t)ph * 32 + c] : ys[(size_t)ph * 16 + (c - 32)];
}

__global__ void gather_feats(float* __restrict__ fm, const float* __restrict__ y,
                             const int* __restrict__ mask, int N, int C) {
  int idx = blockIdx.x * blockDim.x + threadIdx.x;
  if (idx >= N * C) return;
  int n = idx / C, c = idx - n * C;
  fm[idx] = y[(size_t)mask[n] * C + c];
}

// cat[n, kk*(C+3)+{feat,rel}] in bf16; pads tail to Kpad with zeros
__global__ void build_cat(__bf16* __restrict__ cat, const float* __restrict__ fm,
                          const float* __restrict__ coors, const int* __restrict__ knn,
                          int N, int C, int Kpad) {
  int idx = blockIdx.x * blockDim.x + threadIdx.x;
  if (idx >= N * 9) return;
  int n = idx / 9, kk = idx - n * 9;
  int j = knn[n * 9 + kk];
  __bf16* dst = cat + (size_t)n * Kpad + kk * (C + 3);
  const float* src = fm + (size_t)j * C;
  for (int c = 0; c < C; ++c) dst[c] = (__bf16)src[c];
#pragma unroll
  for (int d = 0; d < 3; ++d) dst[C + d] = (__bf16)(coors[n * 3 + d] - coors[j * 3 + d]);
  if (kk == 8) {
    for (int k = 9 * (C + 3); k < Kpad; ++k) cat[(size_t)n * Kpad + k] = (__bf16)0.0f;
  }
}

__global__ void scatter_add(float* __restrict__ s, const float* __restrict__ fm,
                            const int* __restrict__ mask, int N, int C) {
  int idx = blockIdx.x * blockDim.x + threadIdx.x;
  if (idx >= N * C) return;
  int n = idx / C, c = idx - n * C;
  s[(size_t)mask[n] * C + c] += fm[idx];  // mask indices unique -> no conflicts
}

__global__ void upsample2x(float* __restrict__ s, const float* __restrict__ c2, int C,
                           int H, int W) {
  int idx = blockIdx.x * blockDim.x + threadIdx.x;
  if (idx >= H * W * C) return;
  int c = idx % C;
  int p = idx / C;
  int ph = ((p / W) >> 1) * (W >> 1) + ((p % W) >> 1);
  s[idx] = c2[(size_t)ph * C + c];
}

// ---------------------------------------------------------------------------
// Host orchestration
// ---------------------------------------------------------------------------
extern "C" void kernel_launch(void* const* d_in, const int* in_sizes, int n_in,
                              void* d_out, int out_size, void* d_ws, size_t ws_size,
                              hipStream_t stream) {
  (void)in_sizes; (void)n_in; (void)out_size; (void)ws_size;
  constexpr int H = 256, W = 512, P = H * W;       // 131072
  constexpr int Hh = 128, Wh = 256, Ph = Hh * Wh;  // 32768
  constexpr int N = 8192;

  auto F = [&](int i) { return (const float*)d_in[i]; };
  const float* img  = F(0);
  const float* sd   = F(1);
  const int*   mask = (const int*)d_in[2];
  const float* coor = F(3);
  const int*   knn  = (const int*)d_in[4];

  // Workspace layout (bytes, all 256-aligned)
  char* wsb = (char*)d_ws;
  size_t off = 0;
  auto alloc = [&](size_t bytes) { size_t o = off; off += (bytes + 255) & ~(size_t)255; return o; };
  const size_t SZ_FULL = (size_t)P * 64 * 4;
  float*  yA   = (float*)(wsb + alloc(SZ_FULL));
  float*  yB   = (float*)(wsb + alloc(SZ_FULL));
  float*  s    = (float*)(wsb + alloc(SZ_FULL));             // also rgbd pack staging
  __bf16* tdw  = (__bf16*)(wsb + alloc((size_t)P * 64 * 2)); // full-res dw staging (bf16)
  __bf16* thdw = (__bf16*)(wsb + alloc((size_t)Ph * 64 * 2)); // half-res dw staging (bf16)
  float*  c1r  = (float*)(wsb + alloc((size_t)Ph * 64 * 4));
  float*  c2r  = (float*)(wsb + alloc((size_t)Ph * 64 * 4));
  float*  fm0  = (float*)(wsb + alloc((size_t)N * 64 * 4));
  float*  fm1  = (float*)(wsb + alloc((size_t)N * 64 * 4));
  float*  fm2  = (float*)(wsb + alloc((size_t)N * 64 * 4));
  __bf16* cat  = (__bf16*)(wsb + alloc((size_t)N * 608 * 2));
  __bf16* wt   = (__bf16*)(wsb + alloc((size_t)64 * MAX_KPAD * 2));
  float*  ys1  = (float*)(wsb + alloc((size_t)Ph * 16 * 4));
  float*  ys2  = (float*)(wsb + alloc((size_t)Ph * 16 * 4));
  float*  yr1  = (float*)(wsb + alloc((size_t)Ph * 32 * 4));
  float*  yr2  = (float*)(wsb + alloc((size_t)Ph * 32 * 4));

  auto cdiv = [](int a, int b) { return (a + b - 1) / b; };
  const int TB = 256;

  // ---- Stems ----
  direct_conv3x3<<<cdiv(Ph * 16, TB), TB, 0, stream>>>(sd, H, W, 1, 2, F(5), F(6), F(7), F(8),
                                                       ys1, Hh, Wh, 16);
  direct_conv3x3<<<cdiv(Ph * 16, TB), TB, 0, stream>>>(ys1, Hh, Wh, 16, 1, F(9), F(10), F(11),
                                                       F(12), ys2, Hh, Wh, 16);
  pack_rgbd<<<cdiv(P, TB), TB, 0, stream>>>(s, img, sd, P);
  direct_conv3x3<<<cdiv(Ph * 32, TB), TB, 0, stream>>>(s, H, W, 4, 2, F(13), F(14), F(15), F(16),
                                                       yr1, Hh, Wh, 32);
  direct_conv3x3<<<cdiv(Ph * 32, TB), TB, 0, stream>>>(yr1, Hh, Wh, 32, 1, F(17), F(18), F(19),
                                                       F(20), yr2, Hh, Wh, 32);
  upcat<<<cdiv(P * 48, TB), TB, 0, stream>>>(yA, yr2, ys2, H, W);

  // ---- 12 fuse blocks ----
  for (int i = 0; i < 12; ++i) {
    const float* Yin  = (i % 2 == 0) ? yA : yB;
    float*       Yout = (i % 2 == 0) ? yB : yA;
    const int C = (i == 0) ? 48 : 64;
    const int K1 = 9 * (C + 3);
    const int K1pad = (K1 + 31) & ~31;     // 480 / 608
    const int K3pad = (9 * C + 31) & ~31;  // 448 / 576
    const int base = 21 + 30 * i;
    const int nT = (C + 15) / 16;

    // --- 3D branch: gather -> cc1 -> cc2 ---
    gather_feats<<<cdiv(N * C, TB), TB, 0, stream>>>(fm0, Yin, mask, N, C);
    build_cat<<<cdiv(N * 9, TB), TB, 0, stream>>>(cat, fm0, coor, knn, N, C, K1pad);
    cvt_wt_cc<<<cdiv(C * K1pad, TB), TB, 0, stream>>>(wt, F(base + 18), K1, C, K1pad);
    dim3 gcc(N / (16 * WAVES), nT);
    wmma_gemm_rows<<<gcc, 32 * WAVES, 0, stream>>>(cat, K1pad, wt, F(base + 19), F(base + 20),
                                                   F(base + 21), fm0, fm1, C);
    build_cat<<<cdiv(N * 9, TB), TB, 0, stream>>>(cat, fm1, coor, knn, N, C, K1pad);
    cvt_wt_cc<<<cdiv(C * K1pad, TB), TB, 0, stream>>>(wt, F(base + 22), K1, C, K1pad);
    wmma_gemm_rows<<<gcc, 32 * WAVES, 0, stream>>>(cat, K1pad, wt, F(base + 23), F(base + 24),
                                                   F(base + 25), fm1, fm2, C);

    // --- 2D branch ---
    dim3 gph(Ph / (16 * WAVES), nT);
    dim3 gpf(P / (16 * WAVES), nT);
    // c1: dw stride2 + pw
    dwconv3x3<<<cdiv(Ph * 64, TB), TB, 0, stream>>>(Yin, H, W, C, 2, F(base + 0), F(base + 1),
                                                    thdw, Hh, Wh, 64);
    cvt_wt_pw<<<cdiv(C * 64, TB), TB, 0, stream>>>(wt, F(base + 2), C, 64, C);
    wmma_gemm_rows<<<gph, 32 * WAVES, 0, stream>>>(thdw, 64, wt, F(base + 3), F(base + 4),
                                                   F(base + 5), nullptr, c1r, C);
    // c2: dw + pw (half res)
    dwconv3x3<<<cdiv(Ph * 64, TB), TB, 0, stream>>>(c1r, Hh, Wh, C, 1, F(base + 6), F(base + 7),
                                                    thdw, Hh, Wh, 64);
    cvt_wt_pw<<<cdiv(C * 64, TB), TB, 0, stream>>>(wt, F(base + 8), C, 64, C);
    wmma_gemm_rows<<<gph, 32 * WAVES, 0, stream>>>(thdw, 64, wt, F(base + 9), F(base + 10),
                                                   F(base + 11), nullptr, c2r, C);
    // s = upsample2x(c2)
    upsample2x<<<cdiv(P * C, TB), TB, 0, stream>>>(s, c2r, C, H, W);
    // c3: dw + pw (full res), added into s via epilogue residual
    dwconv3x3<<<cdiv(P * 64, TB), TB, 0, stream>>>(Yin, H, W, C, 1, F(base + 12), F(base + 13),
                                                   tdw, H, W, 64);
    cvt_wt_pw<<<cdiv(C * 64, TB), TB, 0, stream>>>(wt, F(base + 14), C, 64, C);
    wmma_gemm_rows<<<gpf, 32 * WAVES, 0, stream>>>(tdw, 64, wt, F(base + 15), F(base + 16),
                                                   F(base + 17), s, s, C);
    // scatter 3D result into s
    scatter_add<<<cdiv(N * C, TB), TB, 0, stream>>>(s, fm2, mask, N, C);

    // --- out conv 3x3 (C -> 64), relu(abn), + residual if i>0 ---
    cvt_wt_conv<<<cdiv(64 * K3pad, TB), TB, 0, stream>>>(wt, F(base + 26), C, K3pad, 64);
    dim3 goc(P / (16 * WAVES), 4);
    wmma_conv3x3<<<goc, 32 * WAVES, 0, stream>>>(s, H, W, C, wt, K3pad, F(base + 27),
                                                 F(base + 28), F(base + 29),
                                                 (i > 0) ? Yin : nullptr, Yout, 64);
  }

  // ---- Output head (final activations in yA after 12 blocks) ----
  const int OB = 21 + 30 * 12;  // 381
  cvt_wt_conv<<<cdiv(32 * 576, TB), TB, 0, stream>>>(wt, F(OB + 0), 64, 576, 32);
  dim3 gh1(P / (16 * WAVES), 2);
  wmma_conv3x3<<<gh1, 32 * WAVES, 0, stream>>>(yA, H, W, 64, wt, 576, F(OB + 1), F(OB + 2),
                                               F(OB + 3), nullptr, s, 32);
  direct_conv3x3<<<cdiv(P, TB), TB, 0, stream>>>(s, H, W, 32, 1, F(OB + 4), F(OB + 5), nullptr,
                                                 nullptr, (float*)d_out, H, W, 1);
}
